// FeedzaiExtraConcatProduction_53223234732117
// MI455X (gfx1250) — matrix operand order repaired and checked by reading, not compiled, not run on hardware
//
#include <hip/hip_runtime.h>

// ---------------------------------------------------------------------------
// Fused gather + GRU cell + scatter + MLP head for MI455X (gfx1250, wave32).
// Matrix math runs on v_wmma_f32_16x16x32_bf16 (fp32 -> bf16 A/B, f32 acc).
// Weights are pre-swizzled into WMMA B-fragment layout in LDS once per block.
// Biases ride inside the WMMA via a constant 1.0 column in the zero-padded
// K range of the A fragments, so all accumulators start from the inline-0
// C operand. fp32->bf16 uses v_cvt_pk_bf16_f32; global offsets are 32-bit
// (GVS form); h transposes go through bank-conflict-free padded LDS.
// ---------------------------------------------------------------------------

typedef __attribute__((ext_vector_type(16))) __bf16 v16bf;
typedef __attribute__((ext_vector_type(2)))  __bf16 v2bf;
typedef __attribute__((ext_vector_type(8)))  float  v8f;
typedef __attribute__((ext_vector_type(8)))  int    v8i;

#define FD   32      // input features
#define UD   48      // GRU units
#define GU   144     // 3*U
#define HD   24      // dense hidden
#define NFRAG 33     // 9 (Wk) + 18 (Uk) + 6 (dense)
#define WAVES 8      // waves per block (256 threads, wave32)
#define HSTR 52      // padded LDS row stride for the 16x48 h tile (words)

// Two f32 -> packed bf16 (one u32 word) in a single VALU op.
__device__ __forceinline__ int cvt2i(float a, float b) {
#if __has_builtin(__builtin_amdgcn_cvt_pk_bf16_f32)
  return __builtin_bit_cast(int, __builtin_amdgcn_cvt_pk_bf16_f32(a, b));
#else
  v2bf r; r[0] = (__bf16)a; r[1] = (__bf16)b;
  return __builtin_bit_cast(int, r);
#endif
}
__device__ __forceinline__ float fast_sigmoid(float x) {
  return __builtin_amdgcn_rcpf(1.0f + __expf(-x));
}
__device__ __forceinline__ float fast_tanh(float x) {
  return 2.0f * __builtin_amdgcn_rcpf(1.0f + __expf(-2.0f * x)) - 1.0f;
}
__device__ __forceinline__ v8f wmma_bf16(v16bf a, v16bf b, v8f c) {
  return __builtin_amdgcn_wmma_f32_16x16x32_bf16(
      /*neg_a=*/false, a, /*neg_b=*/false, b,
      /*c_mod=*/(short)0, c, /*reuse_a=*/false, /*reuse_b=*/false);
}
// Build a 16x32 bf16 A-fragment from four float4 runs.
// Element j of the vector is K = 16*(j>=8) + 8*half + (j&7)  (per CDNA5 ISA
// 16-bit A layout: lanes 0-15 hold K 0-7/16-23, lanes 16-31 hold K 8-15/24-31).
__device__ __forceinline__ v16bf mk_afrag(float4 f0, float4 f1,
                                          float4 f2, float4 f3) {
  v8i w;
  w[0] = cvt2i(f0.x, f0.y); w[1] = cvt2i(f0.z, f0.w);
  w[2] = cvt2i(f1.x, f1.y); w[3] = cvt2i(f1.z, f1.w);
  w[4] = cvt2i(f2.x, f2.y); w[5] = cvt2i(f2.z, f2.w);
  w[6] = cvt2i(f3.x, f3.y); w[7] = cvt2i(f3.z, f3.w);
  return __builtin_bit_cast(v16bf, w);
}
// Half fragment: K 0..15 real (two float4 runs); word 4 carries the constant
// 1.0 column (klocal 16, half-0 lanes) used to inject biases via the B side.
__device__ __forceinline__ v16bf mk_afrag_lo(float4 f0, float4 f1, int oneW) {
  v8i w;
  w[0] = cvt2i(f0.x, f0.y); w[1] = cvt2i(f0.z, f0.w);
  w[2] = cvt2i(f1.x, f1.y); w[3] = cvt2i(f1.z, f1.w);
  w[4] = oneW; w[5] = 0; w[6] = 0; w[7] = 0;
  return __builtin_bit_cast(v16bf, w);
}
__device__ __forceinline__ v16bf load_frag(const unsigned* ldsW, int f, int lane) {
  v8i w = *(const v8i*)(ldsW + f * 256 + lane * 8);   // 32B ds_load
  return __builtin_bit_cast(v16bf, w);
}

__global__ void __launch_bounds__(256)
copy_state_kernel(const float* __restrict__ src, float* __restrict__ dst, int n) {
  int i = blockIdx.x * blockDim.x + threadIdx.x;
  if (i < n) dst[i] = src[i];
}

__global__ void __launch_bounds__(256)
fused_gru_kernel(const float* __restrict__ inputs, const int* __restrict__ ids,
                 const float* __restrict__ state,
                 const float* __restrict__ Wk, const float* __restrict__ Uk,
                 const float* __restrict__ gbias,
                 const float* __restrict__ Dw, const float* __restrict__ Db,
                 const float* __restrict__ Ow, const float* __restrict__ Ob,
                 float* __restrict__ out, float* __restrict__ out_state,
                 int Btot) {
  // Packed bf16 B-fragments: [frag][lane][vgpr] u32 words.
  __shared__ unsigned ldsW[NFRAG * 256];
  __shared__ float ldsBias[GU + HD + 1];        // gru_bias | out_w | out_b
  __shared__ float ldsH[WAVES * 16 * HSTR];     // per-wave h tile (padded rows)

  const int tid = threadIdx.x;

  // ---- One-time per block: pre-swizzle weights into WMMA B-layout --------
  // B-fragment layout (16x16x32 bf16): lane L holds column n = L%16;
  // VGPR v holds K = 8*(L/16) + 2v + 8*(v>=4) (+0/+1 in lo/hi halves).
  // Row klocal==16 (hf==0, v==4, lo half) of zero-padded fragments carries
  // the bias vector, matched by the 1.0 column in the A fragments.
  for (int idx = tid; idx < NFRAG * 256; idx += 256) {
    int f = idx >> 8;
    int l = (idx >> 3) & 31;
    int v = idx & 7;
    int nli = l & 15;
    int hf = l >> 4;
    int k0 = hf * 8 + 2 * v + ((v >> 2) << 3);
    float w0 = 0.f, w1 = 0.f;
    if (f < 9) {                                  // Wk tiles: t = f, K = 0..31
      int n = f * 16 + nli;
      w0 = Wk[(k0 + 0) * GU + n];
      w1 = Wk[(k0 + 1) * GU + n];
    } else if (f < 27) {                          // Uk: t = g>>1, kstep = g&1
      int g = f - 9, t = g >> 1, s = g & 1;
      int n = t * 16 + nli;
      int ka = 32 * s + k0;
      w0 = (ka + 0 < UD) ? Uk[(ka + 0) * GU + n] : 0.f;
      w1 = (ka + 1 < UD) ? Uk[(ka + 1) * GU + n] : 0.f;
      // z/r gate bias rides at klocal 16 of the kstep-1 fragment.
      if (s == 1 && hf == 0 && v == 4 && t < 6) w0 = gbias[n];
    } else {                                      // dense: t = g/3, kstep = g%3
      int g = f - 27, t = g / 3, s = g % 3;
      int n = t * 16 + nli;
      int ka = 32 * s + k0;
      bool okn = n < HD;
      w0 = (okn && (ka + 0) < (FD + UD)) ? Dw[(ka + 0) * HD + n] : 0.f;
      w1 = (okn && (ka + 1) < (FD + UD)) ? Dw[(ka + 1) * HD + n] : 0.f;
      // dense bias rides at klocal 16 of the kstep-2 fragment.
      if (s == 2 && hf == 0 && v == 4 && okn) w0 = Db[n];
    }
    ldsW[idx] = (unsigned)cvt2i(w0, w1);
  }
  for (int i = tid; i < GU; i += 256) ldsBias[i] = gbias[i];
  for (int i = tid; i < HD; i += 256) ldsBias[GU + i] = Ow[i];
  if (tid == 0) ldsBias[GU + HD] = Ob[0];
  __syncthreads();

  const int lane = tid & 31;
  const int wave = tid >> 5;
  const int laneHalf = lane >> 4;   // 0 / 1
  const int nl = lane & 15;         // N column (B/C layout) or M row (A layout)
  float* myH = &ldsH[wave * 16 * HSTR];
  // bf16 1.0 in the low half-word: A element j=8 (klocal 16) on half-0 lanes.
  const int oneW = laneHalf ? 0 : 0x00003F80;

  const int numTiles = (Btot + 15) >> 4;
  const int tileStride = gridDim.x * WAVES;
  for (int tile = blockIdx.x * WAVES + wave; tile < numTiles;
       tile += tileStride) {
    const int rowBase = tile * 16;

    // ---- A-layout row for this lane (M = nl; halves load the same row) ----
    int rowA = rowBase + nl;
    int rowAc = rowA < Btot ? rowA : Btot - 1;
    int idA = ids[rowAc];

    // Prefetch next tile's input row for this lane (global_prefetch_b8).
    {
      int nrow = rowA + tileStride * 16;
      if (nrow < Btot) __builtin_prefetch(inputs + (unsigned)nrow * FD, 0, 0);
    }

    // A fragment of x (16x32, K = all 32 features); 32-bit GVS offsets.
    const unsigned xoff = (unsigned)rowAc * FD + 8u * laneHalf;
    v16bf Ax = mk_afrag(*(const float4*)(inputs + xoff),
                        *(const float4*)(inputs + xoff + 4),
                        *(const float4*)(inputs + xoff + 16),
                        *(const float4*)(inputs + xoff + 20));

    // h_prev slice for this lane (row nl, three 8-col runs), kept as f32
    // for the LDS transpose and converted to bf16 for the A fragments.
    const unsigned hoff = (unsigned)idA * UD + 8u * laneHalf;
    float4 h0 = *(const float4*)(state + hoff);
    float4 h1 = *(const float4*)(state + hoff + 4);
    float4 h2 = *(const float4*)(state + hoff + 16);
    float4 h3 = *(const float4*)(state + hoff + 20);
    float4 h4 = *(const float4*)(state + hoff + 32);
    float4 h5 = *(const float4*)(state + hoff + 36);
    v16bf Ah0 = mk_afrag(h0, h1, h2, h3);
    v16bf Ah1 = mk_afrag_lo(h4, h5, oneW);      // + 1.0 bias column

    // Stage h_prev into the per-wave LDS tile (row nl, A-layout col runs).
    {
      float* hpr = myH + nl * HSTR + 8 * laneHalf;
      *(float4*)(hpr)      = h0;  *(float4*)(hpr + 4)  = h1;
      *(float4*)(hpr + 16) = h2;  *(float4*)(hpr + 20) = h3;
      *(float4*)(hpr + 32) = h4;  *(float4*)(hpr + 36) = h5;
    }

    // C-layout ids via cross-lane broadcast (lane r holds ids[rowBase+r]).
    unsigned soff[8];
#pragma unroll
    for (int r = 0; r < 8; ++r)
      soff[r] = (unsigned)__shfl(idA, r + 8 * laneHalf, 32) * UD;

    // All lanes' h_prev stores must land before the transposed reads.
    asm volatile("s_wait_dscnt 0x0" ::: "memory");

    // ---- GRU gates: 3 column tiles of 16 over U=48 ----
#pragma unroll
    for (int t = 0; t < 3; ++t) {
      float bh = ldsBias[96 + t * 16 + nl];      // candidate bias (elementwise)
      const v8f zero = {};
      v8f accZ, accR, accX, accH;
      // z gate (gate tile t): x@Wk + h@Uk + bias (bias row in kstep-1 frag)
      accZ = wmma_bf16(Ax,  load_frag(ldsW, t, lane), zero);
      accZ = wmma_bf16(Ah0, load_frag(ldsW, 9 + 2 * t, lane), accZ);
      accZ = wmma_bf16(Ah1, load_frag(ldsW, 9 + 2 * t + 1, lane), accZ);
      // r gate (gate tile t+3)
      accR = wmma_bf16(Ax,  load_frag(ldsW, 3 + t, lane), zero);
      accR = wmma_bf16(Ah0, load_frag(ldsW, 9 + 2 * (3 + t), lane), accR);
      accR = wmma_bf16(Ah1, load_frag(ldsW, 9 + 2 * (3 + t) + 1, lane), accR);
      // candidate: xh and hh kept separate (hcand = tanh(xh + bh + r*hh))
      accX = wmma_bf16(Ax,  load_frag(ldsW, 6 + t, lane), zero);
      accH = wmma_bf16(Ah0, load_frag(ldsW, 9 + 2 * (6 + t), lane), zero);
      accH = wmma_bf16(Ah1, load_frag(ldsW, 9 + 2 * (6 + t) + 1, lane), accH);

      const int ncol = t * 16 + nl;
#pragma unroll
      for (int r = 0; r < 8; ++r) {
        int m = r + 8 * laneHalf;
        // h_prev from LDS: one address VGPR + constant immediate offsets.
        float hp = myH[m * HSTR + ncol];
        float z  = fast_sigmoid(accZ[r]);
        float rr = fast_sigmoid(accR[r]);
        float hc = fast_tanh((accX[r] + bh) + rr * accH[r]);
        float hn = z * hp + (1.f - z) * hc;
        myH[m * HSTR + ncol] = hn;               // overwrite tile with h_new
        int rowC = rowBase + m;
        if (rowC < Btot)
          out_state[soff[r] + (unsigned)ncol] = hn;   // scatter (GVS form)
      }
    }

    // h_new stores must land before the transposed dense-GEMM reads.
    asm volatile("s_wait_dscnt 0x0" ::: "memory");

    // ---- dense head: feat = [x | h_new], K = 80 -> 3 K-steps ----
    const float* hrow = myH + nl * HSTR + 8 * laneHalf;  // A-layout row M = nl
    v16bf Af1 = mk_afrag(*(const float4*)(hrow),
                         *(const float4*)(hrow + 4),
                         *(const float4*)(hrow + 16),
                         *(const float4*)(hrow + 20));
    v16bf Af2 = mk_afrag_lo(*(const float4*)(hrow + 32),
                            *(const float4*)(hrow + 36), oneW);  // + bias col

    v8f hid0, hid1;
#pragma unroll
    for (int t = 0; t < 2; ++t) {
      const v8f zero = {};
      v8f acc;
      acc = wmma_bf16(Ax,  load_frag(ldsW, 27 + t * 3 + 0, lane), zero);
      acc = wmma_bf16(Af1, load_frag(ldsW, 27 + t * 3 + 1, lane), acc);
      acc = wmma_bf16(Af2, load_frag(ldsW, 27 + t * 3 + 2, lane), acc);
#pragma unroll
      for (int r = 0; r < 8; ++r) acc[r] = fmaxf(acc[r], 0.f);  // relu
      if (t == 0) hid0 = acc; else hid1 = acc;
    }

    // ---- output projection (24 -> 1): 16-lane shuffle reduction ----
    float w0 = ldsBias[GU + nl];                               // out_w[nl]
    float w1 = (nl < 8) ? ldsBias[GU + 16 + nl] : 0.f;         // out_w[16+nl]
    float ob = ldsBias[GU + HD];
    float s[8];
#pragma unroll
    for (int r = 0; r < 8; ++r) s[r] = hid0[r] * w0 + hid1[r] * w1;
#pragma unroll
    for (int m = 1; m < 16; m <<= 1) {
#pragma unroll
      for (int r = 0; r < 8; ++r) s[r] += __shfl_xor(s[r], m, 16);
    }
    float val = s[0];
#pragma unroll
    for (int r = 1; r < 8; ++r) val = (nl == r) ? s[r] : val;
    int rowO = rowBase + nl + 8 * laneHalf;                    // M = nl + 8*half
    if (nl < 8 && rowO < Btot)
      out[(unsigned)rowO] = fast_sigmoid(val + ob);
  }
}

extern "C" void kernel_launch(void* const* d_in, const int* in_sizes, int n_in,
                              void* d_out, int out_size, void* d_ws, size_t ws_size,
                              hipStream_t stream) {
  const float* inputs = (const float*)d_in[0];
  const int*   ids    = (const int*)d_in[1];
  const float* state  = (const float*)d_in[2];
  const float* Wk     = (const float*)d_in[3];
  const float* Uk     = (const float*)d_in[4];
  const float* gbias  = (const float*)d_in[5];
  const float* Dw     = (const float*)d_in[6];
  const float* Db     = (const float*)d_in[7];
  const float* Ow     = (const float*)d_in[8];
  const float* Ob     = (const float*)d_in[9];

  const int Btot = in_sizes[1];          // 500,000
  const int SU   = in_sizes[2];          // 1000 * 48

  float* out       = (float*)d_out;
  float* out_state = out + Btot;

  // 1) new_state starts as a copy of state (scatter overwrites updated rows)
  copy_state_kernel<<<(SU + 255) / 256, 256, 0, stream>>>(state, out_state, SU);

  // 2) fused gather + GRU + scatter + MLP head
  int numTiles = (Btot + 15) / 16;
  int blocks = (numTiles + WAVES - 1) / WAVES;
  if (blocks > 2048) blocks = 2048;
  fused_gru_kernel<<<blocks, 256, 0, stream>>>(
      inputs, ids, state, Wk, Uk, gbias, Dw, Db, Ow, Ob,
      out, out_state, Btot);
}
